// MogLSTM_BiDir_11020886081719
// MI455X (gfx1250) — matrix operand order repaired and checked by reading, not compiled, not run on hardware
//
#include <hip/hip_runtime.h>
#include <hip/hip_bf16.h>
#include <cstdint>
#include <cstddef>

// ---------------------------------------------------------------------------
// MogLSTM BiDir for MI455X (gfx1250, wave32, WMMA).
// Strategy: bf16 weights (L2-resident, 105MB total), fp32 state, WMMA
// v_wmma_f32_16x16x32_bf16 for every matmul, fused sigmoid/tanh epilogues.
// ---------------------------------------------------------------------------

typedef __bf16 bf16_t;
typedef __attribute__((ext_vector_type(16))) __bf16 v16bf;
typedef __attribute__((ext_vector_type(8)))  __bf16 v8bf;
typedef __attribute__((ext_vector_type(8)))  float  v8f;

#define DEV __device__ __forceinline__

DEV float sigmf(float x) { return 1.0f / (1.0f + __expf(-x)); }

// ---- WMMA fragment loaders (layouts per CDNA5 ISA §7.12.2) -----------------
// A (16x32, MxK, bf16, row-major, leading dim lda):
//   lanes 0-15 : row = m0+lane, K = k0+{0..7} and k0+{16..23}
//   lanes 16-31: row = m0+lane-16, K = k0+{8..15} and k0+{24..31}
DEV v16bf load_frag_A(const bf16_t* __restrict__ A, int lda, int m0, int k0, int lane) {
    const int row = m0 + (lane & 15);
    const int kb  = k0 + ((lane >> 4) << 3);
    const bf16_t* p = A + (size_t)row * lda + kb;
    v8bf lo = *reinterpret_cast<const v8bf*>(p);
    v8bf hi = *reinterpret_cast<const v8bf*>(p + 16);
    v16bf f;
#pragma unroll
    for (int i = 0; i < 8; ++i) { f[i] = lo[i]; f[i + 8] = hi[i]; }
    return f;
}

// B (32x16, KxN) for C = A * W^T with W[N][K] row-major:
//   element (k, n) = W[n0+n][k].  lane = column (lane&15),
//   lanes 0-15 hold K = k0+{0..7,16..23}, lanes 16-31 hold K = k0+{8..15,24..31}.
//   -> each lane reads two 8-bf16 (16B) chunks along K of one W row.
DEV v16bf load_frag_B(const bf16_t* __restrict__ W, int ldw, int n0, int k0, int lane) {
    const int row = n0 + (lane & 15);
    const int kb  = k0 + ((lane >> 4) << 3);
    const bf16_t* p = W + (size_t)row * ldw + kb;
    v8bf lo = *reinterpret_cast<const v8bf*>(p);
    v8bf hi = *reinterpret_cast<const v8bf*>(p + 16);
    v16bf f;
#pragma unroll
    for (int i = 0; i < 8; ++i) { f[i] = lo[i]; f[i + 8] = hi[i]; }
    return f;
}

// ---- GEMM: OUT[64, N] = A[64, K] * W[N, K]^T (+bias) with fused epilogue ----
// One wave per (16-wide N tile, 32-wide M half). M=64 fixed.
//   EPI 0 (MOG)  : t = 2*sigmoid(acc+bias)*outF; outF = t; outB = bf16(t)
//   EPI 1 (STORE): outF = acc
enum { EPI_MOG = 0, EPI_STORE = 1 };

template <int EPI>
__global__ __launch_bounds__(256) void gemm_bf16_wmma(
    const bf16_t* __restrict__ A, int lda,
    const bf16_t* __restrict__ W, int ldw,
    const float* __restrict__ bias,
    int N, int K,
    float* __restrict__ outF, int ldo,
    bf16_t* __restrict__ outB)
{
    const int wid  = (int)((blockIdx.x * blockDim.x + threadIdx.x) >> 5);
    const int lane = (int)(threadIdx.x & 31);
    const int n0   = (wid >> 1) * 16;
    const int m0   = (wid & 1) * 32;
    if (n0 >= N) return;  // wave-uniform: EXEC stays all-ones for WMMA

    v8f acc0 = {};
    v8f acc1 = {};
    for (int k0 = 0; k0 < K; k0 += 32) {
        v16bf fb  = load_frag_B(W, ldw, n0, k0, lane);
        v16bf fa0 = load_frag_A(A, lda, m0,      k0, lane);
        v16bf fa1 = load_frag_A(A, lda, m0 + 16, k0, lane);
        acc0 = __builtin_amdgcn_wmma_f32_16x16x32_bf16(false, fa0, false, fb, (short)0, acc0, false, false);
        acc1 = __builtin_amdgcn_wmma_f32_16x16x32_bf16(false, fa1, false, fb, (short)0, acc1, false, false);
    }

    // C/D layout: acc[j] <-> row = mbase + (lane<16 ? j : 8+j), col = n0 + (lane&15)
    const int   col   = n0 + (lane & 15);
    const float bv    = bias ? bias[col] : 0.0f;
    const int   rbase = m0 + ((lane >> 4) << 3);
#pragma unroll
    for (int j = 0; j < 8; ++j) {
        const int r0 = rbase + j;
        const int r1 = r0 + 16;
        float v0 = acc0[j] + bv;
        float v1 = acc1[j] + bv;
        if (EPI == EPI_MOG) {
            const size_t i0 = (size_t)r0 * ldo + col;
            const size_t i1 = (size_t)r1 * ldo + col;
            float t0 = (2.0f * sigmf(v0)) * outF[i0];
            float t1 = (2.0f * sigmf(v1)) * outF[i1];
            outF[i0] = t0; outB[i0] = (bf16_t)t0;
            outF[i1] = t1; outB[i1] = (bf16_t)t1;
        } else {
            outF[(size_t)r0 * ldo + col] = v0;
            outF[(size_t)r1 * ldo + col] = v1;
        }
    }
}

// ---- fused LSTM pointwise update (gate order i,f,g,o) ----------------------
// NEXT 0: forward h into next cell's x region; 1: store h to out; 2: out += h
template <int NEXT>
__global__ __launch_bounds__(256) void lstm_cell_update(
    const float* __restrict__ gates,  // [64][4096]
    const float* __restrict__ biasG,  // [4096] (= b_ih + b_hh)
    float* __restrict__ cst,          // [64][1024]
    float* __restrict__ xhF, bf16_t* __restrict__ xhB,  // own XH, h at col 1024
    float* __restrict__ nxF, bf16_t* __restrict__ nxB,  // next cell's x region
    float* __restrict__ outp)                           // output slice [64][1024]
{
    const int i = (int)(blockIdx.x * blockDim.x + threadIdx.x);  // 65536
    const int b = i >> 10, j = i & 1023;
    const float* g = gates + (size_t)b * 4096;
    const float ig = sigmf(g[j]        + biasG[j]);
    const float fg = sigmf(g[1024 + j] + biasG[1024 + j]);
    const float gg = tanhf(g[2048 + j] + biasG[2048 + j]);
    const float og = sigmf(g[3072 + j] + biasG[3072 + j]);
    const float cn = fg * cst[i] + ig * gg;
    const float hn = og * tanhf(cn);
    cst[i] = cn;
    const size_t hi = (size_t)b * 2048 + 1024 + j;
    xhF[hi] = hn; xhB[hi] = (bf16_t)hn;
    if (NEXT == 0)      { const size_t xi = (size_t)b * 2048 + j; nxF[xi] = hn; nxB[xi] = (bf16_t)hn; }
    else if (NEXT == 1) { outp[i] = hn; }
    else                { outp[i] += hn; }
}

// ---- small utility kernels -------------------------------------------------
__global__ __launch_bounds__(256) void stage_x_kernel(
    const float* __restrict__ src, float* __restrict__ xF, bf16_t* __restrict__ xB)
{
    const int i = (int)(blockIdx.x * blockDim.x + threadIdx.x);  // 65536
    const int b = i >> 10, j = i & 1023;
    const float v = src[i];
    xF[(size_t)b * 2048 + j] = v;
    xB[(size_t)b * 2048 + j] = (bf16_t)v;
}

__global__ __launch_bounds__(256) void f32_to_bf16_kernel(
    const float* __restrict__ src, bf16_t* __restrict__ dst, int n)
{
    const int i = (int)(blockIdx.x * blockDim.x + threadIdx.x);
    if (i < n) dst[i] = (bf16_t)src[i];
}

// Wcat[4096][2048]: cols [0,1024) = w_ih row, cols [1024,2048) = w_hh row
__global__ __launch_bounds__(256) void pack_wcat_kernel(
    const float* __restrict__ wih, const float* __restrict__ whh, bf16_t* __restrict__ wcat)
{
    const size_t i = (size_t)blockIdx.x * blockDim.x + threadIdx.x;  // 4096*2048
    const int n = (int)(i >> 11), k = (int)(i & 2047);
    const float v = (k < 1024) ? wih[(size_t)n * 1024 + k] : whh[(size_t)n * 1024 + (k - 1024)];
    wcat[i] = (bf16_t)v;
}

__global__ __launch_bounds__(256) void add_vec_kernel(
    const float* __restrict__ a, const float* __restrict__ b, float* __restrict__ o, int n)
{
    const int i = (int)(blockIdx.x * blockDim.x + threadIdx.x);
    if (i < n) o[i] = a[i] + b[i];
}

__global__ __launch_bounds__(256) void zero_u32_kernel(uint32_t* __restrict__ p, int n)
{
    const int i = (int)(blockIdx.x * blockDim.x + threadIdx.x);
    if (i < n) p[i] = 0u;
}

__global__ __launch_bounds__(256) void copy_state_kernel(
    const float* __restrict__ xhF, const float* __restrict__ cst,
    float* __restrict__ hDst, float* __restrict__ cDst)
{
    const int i = (int)(blockIdx.x * blockDim.x + threadIdx.x);  // 65536
    const int b = i >> 10, j = i & 1023;
    hDst[i] = xhF[(size_t)b * 2048 + 1024 + j];
    cDst[i] = cst[i];
}

// ---------------------------------------------------------------------------
extern "C" void kernel_launch(void* const* d_in, const int* in_sizes, int n_in,
                              void* d_out, int out_size, void* d_ws, size_t ws_size,
                              hipStream_t stream)
{
    (void)n_in; (void)out_size; (void)ws_size;
    constexpr int S = 512, B = 64, I = 1024, H = 1024;

    const float* input_seq = (const float*)d_in[0];

    // ---- map params (robust to insertion-order vs sorted-pytree order) ----
    // Per-cell leaves, insertion order: (mog_w,mog_b)x5, w_ih, w_hh, b_ih, b_hh
    // sorted-key order:                 b_hh, b_ih, (mog_w,mog_b)x5, w_hh, w_ih
    const bool sorted_keys = (in_sizes[1] == 4096);
    // logical cell index: 0=fwd L1, 1=fwd L2, 2=bwd L1, 3=bwd L2
    int slot_of[4];
    if (sorted_keys) { slot_of[0] = 2; slot_of[1] = 3; slot_of[2] = 0; slot_of[3] = 1; }
    else             { slot_of[0] = 0; slot_of[1] = 1; slot_of[2] = 2; slot_of[3] = 3; }

    const float *mogW[4][5], *mogB[4][5], *wih[4], *whh[4], *bih[4], *bhh[4];
    for (int c = 0; c < 4; ++c) {
        const int base = 1 + slot_of[c] * 14;
        if (sorted_keys) {
            bhh[c] = (const float*)d_in[base + 0];
            bih[c] = (const float*)d_in[base + 1];
            for (int i = 0; i < 5; ++i) {
                mogW[c][i] = (const float*)d_in[base + 2 + 2 * i];
                mogB[c][i] = (const float*)d_in[base + 3 + 2 * i];
            }
            whh[c] = (const float*)d_in[base + 12];
            wih[c] = (const float*)d_in[base + 13];
        } else {
            for (int i = 0; i < 5; ++i) {
                mogW[c][i] = (const float*)d_in[base + 2 * i];
                mogB[c][i] = (const float*)d_in[base + 2 * i + 1];
            }
            wih[c] = (const float*)d_in[base + 10];
            whh[c] = (const float*)d_in[base + 11];
            bih[c] = (const float*)d_in[base + 12];
            bhh[c] = (const float*)d_in[base + 13];
        }
    }

    // ---- workspace carve-out ----
    size_t off = 0;
    auto alloc = [&](size_t bytes) -> void* {
        off = (off + 255) & ~(size_t)255;
        void* p = (char*)d_ws + off;
        off += bytes;
        return p;
    };
    bf16_t* wm[4][5]; bf16_t* wcat[4]; float* bg[4];
    float* xhF[4]; bf16_t* xhB[4]; float* cst[4];
    for (int c = 0; c < 4; ++c) {
        for (int i = 0; i < 5; ++i) wm[c][i] = (bf16_t*)alloc((size_t)1024 * 1024 * 2);
        wcat[c] = (bf16_t*)alloc((size_t)4096 * 2048 * 2);
        bg[c]   = (float*)alloc((size_t)4096 * 4);
        xhF[c]  = (float*)alloc((size_t)B * 2048 * 4);
        xhB[c]  = (bf16_t*)alloc((size_t)B * 2048 * 2);
        cst[c]  = (float*)alloc((size_t)B * 1024 * 4);
    }
    float* gates = (float*)alloc((size_t)B * 4096 * 4);

    // ---- one-time (per call) weight conversion + state zeroing ----
    for (int c = 0; c < 4; ++c) {
        for (int i = 0; i < 5; ++i)
            f32_to_bf16_kernel<<<(1024 * 1024) / 256, 256, 0, stream>>>(mogW[c][i], wm[c][i], 1024 * 1024);
        pack_wcat_kernel<<<(4096 * 2048) / 256, 256, 0, stream>>>(wih[c], whh[c], wcat[c]);
        add_vec_kernel<<<16, 256, 0, stream>>>(bih[c], bhh[c], bg[c], 4096);
        zero_u32_kernel<<<(B * 2048) / 256, 256, 0, stream>>>((uint32_t*)xhF[c], B * 2048);
        zero_u32_kernel<<<(B * 1024) / 256, 256, 0, stream>>>((uint32_t*)xhB[c], B * 1024);  // bf16: 2/word
        zero_u32_kernel<<<(B * 1024) / 256, 256, 0, stream>>>((uint32_t*)cst[c], B * 1024);
    }

    float* outSeq = (float*)d_out;                       // [S][B][H]
    float* lastH  = outSeq + (size_t)S * B * H;          // [4][B][H]
    float* lastC  = lastH + (size_t)4 * B * H;           // [4][B][H]

    auto run_cell = [&](int c, int nextMode, float* nxF, bf16_t* nxB, float* outp) {
        // 5 mogrifier steps (even i: x = 2σ(h·Wᵀ+b)·x ; odd i: h = 2σ(x·Wᵀ+b)·h)
        for (int i = 0; i < 5; ++i) {
            const bf16_t* Ap; float* tF; bf16_t* tB;
            if ((i & 1) == 0) { Ap = xhB[c] + 1024; tF = xhF[c];        tB = xhB[c];        }
            else              { Ap = xhB[c];        tF = xhF[c] + 1024; tB = xhB[c] + 1024; }
            gemm_bf16_wmma<EPI_MOG><<<16, 256, 0, stream>>>(
                Ap, 2048, wm[c][i], 1024, mogB[c][i], 1024, 1024, tF, 2048, tB);
        }
        // gates = [x|h] · Wcatᵀ  (K=2048, N=4096)
        gemm_bf16_wmma<EPI_STORE><<<64, 256, 0, stream>>>(
            xhB[c], 2048, wcat[c], 2048, nullptr, 4096, 2048, gates, 4096, nullptr);
        if (nextMode == 0)
            lstm_cell_update<0><<<256, 256, 0, stream>>>(gates, bg[c], cst[c], xhF[c], xhB[c], nxF, nxB, nullptr);
        else if (nextMode == 1)
            lstm_cell_update<1><<<256, 256, 0, stream>>>(gates, bg[c], cst[c], xhF[c], xhB[c], nullptr, nullptr, outp);
        else
            lstm_cell_update<2><<<256, 256, 0, stream>>>(gates, bg[c], cst[c], xhF[c], xhB[c], nullptr, nullptr, outp);
    };

    for (int dir = 0; dir < 2; ++dir) {
        const int c1 = dir * 2 + 0, c2 = dir * 2 + 1;
        for (int t = 0; t < S; ++t) {
            const int ts = (dir == 0) ? t : (S - 1 - t);
            stage_x_kernel<<<(B * I) / 256, 256, 0, stream>>>(
                input_seq + (size_t)ts * B * I, xhF[c1], xhB[c1]);
            run_cell(c1, /*nextMode=*/0, xhF[c2], xhB[c2], nullptr);
            run_cell(c2, /*nextMode=*/(dir == 0) ? 1 : 2, nullptr, nullptr,
                     outSeq + (size_t)t * B * H);
        }
    }

    // last_hidden stack order: [fh1, bh1, fh2, bh2]; same for cells
    const int stackCell[4] = {0, 2, 1, 3};
    for (int s = 0; s < 4; ++s) {
        const int c = stackCell[s];
        copy_state_kernel<<<(B * H) / 256, 256, 0, stream>>>(
            xhF[c], cst[c], lastH + (size_t)s * B * H, lastC + (size_t)s * B * H);
    }
}
